// DOLPHIN_ENCODER_39307540693947
// MI455X (gfx1250) — matrix operand order repaired and checked by reading, not compiled, not run on hardware
//
// MI455X (gfx1250) Conformer encoder: f16-WMMA GEMM engine + fused elementwise ops.
// Roofline: ~330 GFLOP of GEMMs vs ~380MB weight traffic (16us @ 23.3TB/s) =>
// matrix-core bound. v_wmma_f32_16x16x32_f16, f32 accumulate, 2x2 register
// tiling per wave (4 WMMAs/stage), double-buffered LDS staging with on-the-fly
// f32->f16 conversion, global_prefetch_b8 for the next K tile.

#include <hip/hip_runtime.h>

typedef _Float16 h16x16 __attribute__((ext_vector_type(16)));
typedef float f32x8 __attribute__((ext_vector_type(8)));

namespace {
constexpr int Lc = 12, Dc = 512, Fc = 2048, Uc = 2048, Kc = 31, KFc = 3,
              Tc = 1024, Hc = 8, LDEc = 6;
constexpr int DKc = Dc / Hc;   // 64
constexpr int U2c = Uc / 2;    // 1024
constexpr int Pc  = 2 * Tc - 1; // 2047
}

// ---------------------------------------------------------------------------
// Generic WMMA GEMM: C[M,N] = act(A[M,K] @ B + bias)
//   A: f32 row-major, lda
//   B: f32, if transB==0 layout (K,N) ldb, else (N,K) ldb
//   C: f32 row-major, ldc;  bias: per-column (may be null)
//   act: 0=none 1=silu 2=gelu(erf)
// Block tile 64(M) x 128(N), K-step 32. 256 threads = 8 waves (2M x 4N);
// each wave owns a 32x32 tile = 2x2 accumulators -> 4 WMMAs per K-step.
// Double-buffered LDS: stage tile k+1 while computing tile k.
// ---------------------------------------------------------------------------
#define GBM 64
#define GBN 128
#define GBK 32

__global__ __launch_bounds__(256) void dk_gemm(
    int M, int N, int Kd,
    const float* __restrict__ A, int lda,
    const float* __restrict__ B, int ldb, int transB,
    float* __restrict__ C, int ldc,
    const float* __restrict__ bias, int act)
{
  __shared__ _Float16 As[2][GBM][GBK + 2];
  __shared__ _Float16 Bs[2][GBN][GBK + 2];   // [n][k]: contiguous frag reads

  const int tid  = threadIdx.x;
  const int lane = tid & 31;
  const int wave = tid >> 5;
  const int wm   = wave & 1;   // 0..1 -> 32-row M sub-tile
  const int wn   = wave >> 1;  // 0..3 -> 32-col N sub-tile
  const int m0   = blockIdx.y * GBM;
  const int n0   = blockIdx.x * GBN;
  const int kh   = lane >> 4;  // lane half
  const int l16  = lane & 15;

  f32x8 acc[2][2] = {};

  // A staging: 2 float4 per thread (64 rows x 32 K)
  const int a_r0 = tid >> 3;        // 0..31 (+32)
  const int a_c  = (tid & 7) * 4;   // 0..28

  auto stage = [&](int bufi, int k0) {
    // ---- A tile (f32 -> f16) ----
#pragma unroll
    for (int i = 0; i < 2; ++i) {
      const int r  = a_r0 + 32 * i;
      const int gr = m0 + r;
      float4 av = {0.f, 0.f, 0.f, 0.f};
      if (gr < M) {
        av = *(const float4*)(A + (size_t)gr * lda + k0 + a_c);
        if (k0 + GBK < Kd)
          __builtin_prefetch(A + (size_t)gr * lda + k0 + GBK + a_c, 0, 3);
      }
      As[bufi][r][a_c + 0] = (_Float16)av.x;
      As[bufi][r][a_c + 1] = (_Float16)av.y;
      As[bufi][r][a_c + 2] = (_Float16)av.z;
      As[bufi][r][a_c + 3] = (_Float16)av.w;
    }
    // ---- B tile (f32 -> f16) ----
    if (transB) {
      // (N,K) layout: contiguous along K -> float4 along K
#pragma unroll
      for (int i = 0; i < 4; ++i) {
        const int idx = tid + 256 * i;     // 0..1023
        const int n   = idx >> 3;          // 0..127
        const int k4  = (idx & 7) * 4;     // 0..28
        const int gn  = n0 + n;
        float4 bv4 = {0.f, 0.f, 0.f, 0.f};
        if (gn < N) bv4 = *(const float4*)(B + (size_t)gn * ldb + k0 + k4);
        Bs[bufi][n][k4 + 0] = (_Float16)bv4.x;
        Bs[bufi][n][k4 + 1] = (_Float16)bv4.y;
        Bs[bufi][n][k4 + 2] = (_Float16)bv4.z;
        Bs[bufi][n][k4 + 3] = (_Float16)bv4.w;
      }
    } else {
      // (K,N) layout: contiguous along N -> float4 along N, scatter to [n][k]
#pragma unroll
      for (int i = 0; i < 4; ++i) {
        const int idx = tid + 256 * i;
        const int nn  = (idx & 31) * 4;    // 0..124
        const int kk  = idx >> 5;          // 0..31
        const int gn  = n0 + nn;
        float4 bv4 = {0.f, 0.f, 0.f, 0.f};
        if (gn + 3 < N) bv4 = *(const float4*)(B + (size_t)(k0 + kk) * ldb + gn);
        Bs[bufi][nn + 0][kk] = (_Float16)bv4.x;
        Bs[bufi][nn + 1][kk] = (_Float16)bv4.y;
        Bs[bufi][nn + 2][kk] = (_Float16)bv4.z;
        Bs[bufi][nn + 3][kk] = (_Float16)bv4.w;
      }
    }
  };

  const int nk = Kd >> 5;   // Kd is always a multiple of 32 here
  stage(0, 0);
  for (int it = 0; it < nk; ++it) {
    __syncthreads();
    const int bi = it & 1;
    if (it + 1 < nk) stage(bi ^ 1, (it + 1) << 5);

    // ---- fragments per ISA 16-bit layouts ----
    h16x16 afr[2], bfr[2];
#pragma unroll
    for (int mi = 0; mi < 2; ++mi) {
      const _Float16* ap = As[bi][(wm << 5) + (mi << 4) + l16];
#pragma unroll
      for (int v = 0; v < 8; ++v) {
        const int ka = ((v >> 2) << 4) + (kh << 3) + ((v & 3) << 1);
        afr[mi][2 * v]     = ap[ka];
        afr[mi][2 * v + 1] = ap[ka + 1];
      }
    }
#pragma unroll
    for (int ni = 0; ni < 2; ++ni) {
      const _Float16* bp = Bs[bi][(wn << 5) + (ni << 4) + l16];
#pragma unroll
      for (int v = 0; v < 8; ++v) {
        const int kb = ((v >> 2) << 4) + (kh << 3) + ((v & 3) << 1);
        bfr[ni][2 * v]     = bp[kb];
        bfr[ni][2 * v + 1] = bp[kb + 1];
      }
    }
#pragma unroll
    for (int mi = 0; mi < 2; ++mi)
#pragma unroll
      for (int ni = 0; ni < 2; ++ni)
        acc[mi][ni] = __builtin_amdgcn_wmma_f32_16x16x32_f16(
            false, afr[mi], false, bfr[ni], (short)0, acc[mi][ni], false, false);
  }

  // ---- epilogue: bias + activation + store ----
#pragma unroll
  for (int ni = 0; ni < 2; ++ni) {
    const int n_g = n0 + (wn << 5) + (ni << 4) + l16;
    if (n_g >= N) continue;
    const float bval = bias ? bias[n_g] : 0.f;
#pragma unroll
    for (int mi = 0; mi < 2; ++mi) {
#pragma unroll
      for (int r = 0; r < 8; ++r) {
        const int m_g = m0 + (wm << 5) + (mi << 4) + (kh << 3) + r;
        if (m_g < M) {
          float vv = acc[mi][ni][r] + bval;
          if (act == 1)      vv = vv / (1.f + __expf(-vv));                   // SiLU
          else if (act == 2) vv = 0.5f * vv * (1.f + erff(vv * 0.70710678f)); // GELU
          C[(size_t)m_g * ldc + n_g] = vv;
        }
      }
    }
  }
}

// ---------------------------------------------------------------------------
// LayerNorm over C columns; one block per row; supports strided src/dst.
// ---------------------------------------------------------------------------
__global__ __launch_bounds__(256) void dk_ln(
    const float* __restrict__ src, int ss,
    float* __restrict__ dst, int ds2,
    const float* __restrict__ g, const float* __restrict__ b, int C)
{
  __shared__ float red[256];
  const int row = blockIdx.x;
  const float* s = src + (size_t)row * ss;
  float* d = dst + (size_t)row * ds2;
  float sum = 0.f, sq = 0.f;
  for (int c = threadIdx.x; c < C; c += 256) { float x = s[c]; sum += x; sq += x * x; }
  red[threadIdx.x] = sum; __syncthreads();
  for (int off = 128; off > 0; off >>= 1) {
    if (threadIdx.x < off) red[threadIdx.x] += red[threadIdx.x + off];
    __syncthreads();
  }
  const float mean = red[0] / C;
  __syncthreads();
  red[threadIdx.x] = sq; __syncthreads();
  for (int off = 128; off > 0; off >>= 1) {
    if (threadIdx.x < off) red[threadIdx.x] += red[threadIdx.x + off];
    __syncthreads();
  }
  const float var = red[0] / C - mean * mean;
  const float inv = rsqrtf(var + 1e-12f);
  for (int c = threadIdx.x; c < C; c += 256)
    d[c] = (s[c] - mean) * inv * g[c] + b[c];
}

// Softmax in-place over rows of length Tc; one block per row.
__global__ __launch_bounds__(256) void dk_softmax(float* __restrict__ sc)
{
  __shared__ float red[256];
  float* rowp = sc + (size_t)blockIdx.x * Tc;
  float vals[4];
  float mx = -3.0e38f;
#pragma unroll
  for (int i = 0; i < 4; ++i) { vals[i] = rowp[threadIdx.x + 256 * i]; mx = fmaxf(mx, vals[i]); }
  red[threadIdx.x] = mx; __syncthreads();
  for (int off = 128; off > 0; off >>= 1) {
    if (threadIdx.x < off) red[threadIdx.x] = fmaxf(red[threadIdx.x], red[threadIdx.x + off]);
    __syncthreads();
  }
  mx = red[0]; __syncthreads();
  float s = 0.f;
#pragma unroll
  for (int i = 0; i < 4; ++i) { vals[i] = __expf(vals[i] - mx); s += vals[i]; }
  red[threadIdx.x] = s; __syncthreads();
  for (int off = 128; off > 0; off >>= 1) {
    if (threadIdx.x < off) red[threadIdx.x] += red[threadIdx.x + off];
    __syncthreads();
  }
  const float inv = 1.f / red[0];
#pragma unroll
  for (int i = 0; i < 4; ++i) rowp[threadIdx.x + 256 * i] = vals[i] * inv;
}

// scores[t, j] += bd[t, T-1-t+j]  (Transformer-XL rel-shift, fused add)
__global__ void dk_shiftadd(float* __restrict__ sc, const float* __restrict__ bd)
{
  const int t = blockIdx.x;
  const float* brow = bd + (size_t)t * Pc + (Tc - 1 - t);
  float* srow = sc + (size_t)t * Tc;
  for (int j = threadIdx.x; j < Tc; j += 256) srow[j] += brow[j];
}

// dst[i] += a * src[i]
__global__ void dk_axpy(float* __restrict__ dst, const float* __restrict__ src,
                        float a, int n)
{
  int i = blockIdx.x * 256 + threadIdx.x;
  if (i < n) dst[i] += a * src[i];
}

// dst[r,c] = src[r,c] + rowv[c]   (rows x cols, contiguous)
__global__ void dk_addrow(float* __restrict__ dst, const float* __restrict__ src,
                          const float* __restrict__ rowv, int rows, int cols)
{
  int i = blockIdx.x * 256 + threadIdx.x;
  if (i >= rows * cols) return;
  dst[i] = src[i] + rowv[i % cols];
}

// out[r*os+c] = a[r*as+c] * b[r*bs+c]
__global__ void dk_mul(float* __restrict__ out, int os,
                       const float* __restrict__ a, int as,
                       const float* __restrict__ b, int bs,
                       int rows, int cols)
{
  int i = blockIdx.x * 256 + threadIdx.x;
  if (i >= rows * cols) return;
  int r = i / cols, c = i % cols;
  out[(size_t)r * os + c] = a[(size_t)r * as + c] * b[(size_t)r * bs + c];
}

// Depthwise conv over time: out[t,c] = b[c] + sum_k in[t-pad+k, c] * w[c,K]
__global__ void dk_dwconv(const float* __restrict__ in, float* __restrict__ out,
                          const float* __restrict__ w, const float* __restrict__ b,
                          int Tn, int C, int Kn, int pad)
{
  int id = blockIdx.x * 256 + threadIdx.x;
  if (id >= Tn * C) return;
  int c = id % C, t = id / C;
  float s = b[c];
  for (int kk = 0; kk < Kn; ++kk) {
    int tt = t - pad + kk;
    if (tt >= 0 && tt < Tn) s += in[(size_t)tt * C + c] * w[c * Kn + kk];
  }
  out[id] = s;
}

// keys: dst[d*T + t] = src[t*D + d]   ((T,D) -> (H,DK,T) flat == (D,T))
__global__ void dk_keys_tr(const float* __restrict__ src, float* __restrict__ dst)
{
  int id = blockIdx.x * 256 + threadIdx.x;
  if (id >= Tc * Dc) return;
  int d = id % Dc, t = id / Dc;
  dst[(size_t)d * Tc + t] = src[id];
}

// vals: dst[h*T*DK + t*DK + dk] = src[t*D + h*DK + dk]
__global__ void dk_vals_tr(const float* __restrict__ src, float* __restrict__ dst)
{
  int id = blockIdx.x * 256 + threadIdx.x;
  if (id >= Tc * Dc) return;
  int d = id % Dc, t = id / Dc;
  int hh = d / DKc, dk = d % DKc;
  dst[(size_t)hh * Tc * DKc + (size_t)t * DKc + dk] = src[id];
}

// ---------------------------------------------------------------------------
extern "C" void kernel_launch(void* const* d_in, const int* in_sizes, int n_in,
                              void* d_out, int out_size, void* d_ws, size_t ws_size,
                              hipStream_t stream) {
  (void)in_sizes; (void)n_in; (void)out_size; (void)ws_size;

  const float* xin  = (const float*)d_in[0];
  const float* pe   = (const float*)d_in[1];
  const float* lnffm_g = (const float*)d_in[2];
  const float* lnffm_b = (const float*)d_in[3];
  const float* lnmha_g = (const float*)d_in[4];
  const float* lnmha_b = (const float*)d_in[5];
  const float* lnmlp_g = (const float*)d_in[6];
  const float* lnmlp_b = (const float*)d_in[7];
  const float* lnff_g  = (const float*)d_in[8];
  const float* lnff_b  = (const float*)d_in[9];
  const float* lnfin_g = (const float*)d_in[10];
  const float* lnfin_b = (const float*)d_in[11];
  const float* csgu_g  = (const float*)d_in[12];
  const float* csgu_b  = (const float*)d_in[13];
  const float* ffm_w1  = (const float*)d_in[14];
  const float* ffm_b1  = (const float*)d_in[15];
  const float* ffm_w2  = (const float*)d_in[16];
  const float* ffm_b2  = (const float*)d_in[17];
  const float* wq  = (const float*)d_in[18];
  const float* bq  = (const float*)d_in[19];
  const float* wk  = (const float*)d_in[20];
  const float* bk  = (const float*)d_in[21];
  const float* wv  = (const float*)d_in[22];
  const float* bv  = (const float*)d_in[23];
  const float* wp  = (const float*)d_in[24];
  const float* wo  = (const float*)d_in[25];
  const float* bo  = (const float*)d_in[26];
  const float* pbu = (const float*)d_in[27];
  const float* pbv = (const float*)d_in[28];
  const float* cg_w1 = (const float*)d_in[29];
  const float* cg_b1 = (const float*)d_in[30];
  const float* cg_cw = (const float*)d_in[31];
  const float* cg_cb = (const float*)d_in[32];
  const float* cg_w2 = (const float*)d_in[33];
  const float* cg_b2 = (const float*)d_in[34];
  const float* fus_w = (const float*)d_in[35];
  const float* fus_b = (const float*)d_in[36];
  const float* mrg_w = (const float*)d_in[37];
  const float* mrg_b = (const float*)d_in[38];
  const float* ff_w1 = (const float*)d_in[39];
  const float* ff_b1 = (const float*)d_in[40];
  const float* ff_w2 = (const float*)d_in[41];
  const float* ff_b2 = (const float*)d_in[42];
  const float* aft_g = (const float*)d_in[43];
  const float* aft_b = (const float*)d_in[44];
  const float* dec_wk = (const float*)d_in[45];
  const float* dec_bk = (const float*)d_in[46];
  const float* dec_wv = (const float*)d_in[47];
  const float* dec_bv = (const float*)d_in[48];

  float* out = (float*)d_out;

  // ---- workspace carve-up (~65 MB floats) ----
  float* W = (float*)d_ws;
  size_t off = 0;
  auto alloc = [&](size_t n) { float* p = W + off; off += n; return p; };
  const size_t TD = (size_t)Tc * Dc;
  float* hbuf = alloc(TD);
  float* ybuf = alloc(TD);
  float* t1   = alloc((size_t)Tc * Uc);   // FFN / cgMLP hidden (F==U)
  float* t2   = alloc(TD);
  float* qb   = alloc(TD);
  float* kbuf = alloc(TD);
  float* vbuf = alloc(TD);
  float* qu   = alloc(TD);
  float* qv   = alloc(TD);
  float* pproj = alloc((size_t)Pc * Dc);
  float* sc   = alloc((size_t)Tc * Tc);
  float* bdh  = alloc((size_t)Tc * Pc);
  float* xatt = alloc(TD);
  float* xg   = alloc((size_t)Tc * U2c);
  float* xg2  = alloc((size_t)Tc * U2c);
  float* xrg  = alloc((size_t)Tc * U2c);
  float* xc   = alloc((size_t)Tc * 2 * Dc);
  float* xc2  = alloc((size_t)Tc * 2 * Dc);

  auto gemm = [&](const float* A, int lda, const float* B, int ldb, int transB,
                  float* C, int ldc, const float* bias, int act,
                  int M, int N, int K) {
    dim3 g((N + GBN - 1) / GBN, (M + GBM - 1) / GBM);
    dk_gemm<<<g, 256, 0, stream>>>(M, N, K, A, lda, B, ldb, transB, C, ldc, bias, act);
  };
  auto ln = [&](const float* s, int ss, float* d, int ds2,
                const float* g, const float* b, int C, int rows) {
    dk_ln<<<rows, 256, 0, stream>>>(s, ss, d, ds2, g, b, C);
  };
  auto axpy = [&](float* d, const float* s, float a, int n) {
    dk_axpy<<<(n + 255) / 256, 256, 0, stream>>>(d, s, a, n);
  };

  // h = x
  hipMemcpyAsync(hbuf, xin, TD * sizeof(float), hipMemcpyDeviceToDevice, stream);

  for (int l = 0; l < Lc; ++l) {
    // ---- macaron FFN (half residual) ----
    ln(hbuf, Dc, ybuf, Dc, lnffm_g + l * Dc, lnffm_b + l * Dc, Dc, Tc);
    gemm(ybuf, Dc, ffm_w1 + (size_t)l * Dc * Fc, Fc, 0, t1, Fc, ffm_b1 + l * Fc, 1, Tc, Fc, Dc);
    gemm(t1, Fc, ffm_w2 + (size_t)l * Fc * Dc, Dc, 0, t2, Dc, ffm_b2 + l * Dc, 0, Tc, Dc, Fc);
    axpy(hbuf, t2, 0.5f, Tc * Dc);

    // ---- rel-pos multi-head attention ----
    ln(hbuf, Dc, ybuf, Dc, lnmha_g + l * Dc, lnmha_b + l * Dc, Dc, Tc);
    gemm(ybuf, Dc, wq + (size_t)l * Dc * Dc, Dc, 0, qb,   Dc, bq + l * Dc, 0, Tc, Dc, Dc);
    gemm(ybuf, Dc, wk + (size_t)l * Dc * Dc, Dc, 0, kbuf, Dc, bk + l * Dc, 0, Tc, Dc, Dc);
    gemm(ybuf, Dc, wv + (size_t)l * Dc * Dc, Dc, 0, vbuf, Dc, bv + l * Dc, 0, Tc, Dc, Dc);
    gemm(pe, Dc, wp + (size_t)l * Dc * Dc, Dc, 0, pproj, Dc, nullptr, 0, Pc, Dc, Dc);
    dk_addrow<<<(Tc * Dc + 255) / 256, 256, 0, stream>>>(qu, qb, pbu + l * Dc, Tc, Dc);
    dk_addrow<<<(Tc * Dc + 255) / 256, 256, 0, stream>>>(qv, qb, pbv + l * Dc, Tc, Dc);
    for (int hh = 0; hh < Hc; ++hh) {
      gemm(qu + hh * DKc, Dc, kbuf + hh * DKc, Dc, 1, sc, Tc, nullptr, 0, Tc, Tc, DKc);
      gemm(qv + hh * DKc, Dc, pproj + hh * DKc, Dc, 1, bdh, Pc, nullptr, 0, Tc, Pc, DKc);
      dk_shiftadd<<<Tc, 256, 0, stream>>>(sc, bdh);
      dk_softmax<<<Tc, 256, 0, stream>>>(sc);
      gemm(sc, Tc, vbuf + hh * DKc, Dc, 0, xatt + hh * DKc, Dc, nullptr, 0, Tc, DKc, Tc);
    }
    // x1 -> left half of concat buffer
    gemm(xatt, Dc, wo + (size_t)l * Dc * Dc, Dc, 0, xc, 2 * Dc, bo + l * Dc, 0, Tc, Dc, Dc);

    // ---- cgMLP branch ----
    ln(hbuf, Dc, ybuf, Dc, lnmlp_g + l * Dc, lnmlp_b + l * Dc, Dc, Tc);
    gemm(ybuf, Dc, cg_w1 + (size_t)l * Dc * Uc, Uc, 0, t1, Uc, cg_b1 + l * Uc, 2, Tc, Uc, Dc);
    ln(t1 + U2c, Uc, xg, U2c, csgu_g + l * U2c, csgu_b + l * U2c, U2c, Tc);
    dk_dwconv<<<(Tc * U2c + 255) / 256, 256, 0, stream>>>(
        xg, xg2, cg_cw + (size_t)l * U2c * Kc, cg_cb + l * U2c, Tc, U2c, Kc, (Kc - 1) / 2);
    dk_mul<<<(Tc * U2c + 255) / 256, 256, 0, stream>>>(xrg, U2c, t1, Uc, xg2, U2c, Tc, U2c);
    // x2 -> right half of concat buffer
    gemm(xrg, U2c, cg_w2 + (size_t)l * U2c * Dc, Dc, 0, xc + Dc, 2 * Dc, cg_b2 + l * Dc, 0, Tc, Dc, U2c);

    // ---- depthwise fusion + merge ----
    dk_dwconv<<<(Tc * 2 * Dc + 255) / 256, 256, 0, stream>>>(
        xc, xc2, fus_w + (size_t)l * 2 * Dc * KFc, fus_b + (size_t)l * 2 * Dc, Tc, 2 * Dc, KFc, (KFc - 1) / 2);
    axpy(xc, xc2, 1.0f, Tc * 2 * Dc);
    gemm(xc, 2 * Dc, mrg_w + (size_t)l * 2 * Dc * Dc, Dc, 0, t2, Dc, mrg_b + l * Dc, 0, Tc, Dc, 2 * Dc);
    axpy(hbuf, t2, 1.0f, Tc * Dc);

    // ---- trailing FFN (half residual) ----
    ln(hbuf, Dc, ybuf, Dc, lnff_g + l * Dc, lnff_b + l * Dc, Dc, Tc);
    gemm(ybuf, Dc, ff_w1 + (size_t)l * Dc * Fc, Fc, 0, t1, Fc, ff_b1 + l * Fc, 1, Tc, Fc, Dc);
    gemm(t1, Fc, ff_w2 + (size_t)l * Fc * Dc, Dc, 0, t2, Dc, ff_b2 + l * Dc, 0, Tc, Dc, Fc);
    axpy(hbuf, t2, 0.5f, Tc * Dc);

    // ---- final per-layer LN (in-place safe) ----
    ln(hbuf, Dc, hbuf, Dc, lnfin_g + l * Dc, lnfin_b + l * Dc, Dc, Tc);
  }

  // ---- after-encoder LN + decoder K/V projections ----
  ln(hbuf, Dc, hbuf, Dc, aft_g, aft_b, Dc, Tc);
  const size_t keysTotal = (size_t)LDEc * Dc * Tc;
  for (int l = 0; l < LDEc; ++l) {
    gemm(hbuf, Dc, dec_wk + (size_t)l * Dc * Dc, Dc, 0, t2, Dc, dec_bk + l * Dc, 0, Tc, Dc, Dc);
    dk_keys_tr<<<(Tc * Dc + 255) / 256, 256, 0, stream>>>(t2, out + (size_t)l * Dc * Tc);
    gemm(hbuf, Dc, dec_wv + (size_t)l * Dc * Dc, Dc, 0, t2, Dc, dec_bv + l * Dc, 0, Tc, Dc, Dc);
    dk_vals_tr<<<(Tc * Dc + 255) / 256, 256, 0, stream>>>(t2, out + keysTotal + (size_t)l * Dc * Tc);
  }
}
